// GENwoenc_40544491274928
// MI455X (gfx1250) — compile-verified, matrix operand order checked by reading, not run on hardware
//
#include <hip/hip_runtime.h>

typedef __attribute__((ext_vector_type(16))) __bf16 v16bf;
typedef __attribute__((ext_vector_type(8)))  float  v8f;

#define DEVINL __device__ __forceinline__

constexpr int CB  = 2;
constexpr int CNX = 2048;
constexpr int CNQ = 2048;
constexpr int CM  = 8192;
constexpr int CE  = 131072;

DEVINL unsigned short f2bfb(float f) {
  unsigned u = __builtin_bit_cast(unsigned, f);
  u += 0x7fffu + ((u >> 16) & 1u);         // round-to-nearest-even to bf16
  return (unsigned short)(u >> 16);
}

DEVINL v16bf ld_v16bf(const unsigned short* p) { return *(const v16bf*)p; }

#define WMMA_BF16(A,Bv,C) \
  __builtin_amdgcn_wmma_f32_16x16x32_bf16(false,(A),false,(Bv),(short)0,(C),false,false)

// ---------------------------------------------------------------------------
// CDNA5 async copy: per-lane 16B  global -> LDS, tracked by ASYNCcnt.
// lds_addr is the 32-bit LDS byte offset (low 32 bits of a generic LDS ptr,
// per ISA 10.2 aperture truncation).
// ---------------------------------------------------------------------------
DEVINL void async_ld_b128(unsigned lds_addr, const void* gaddr) {
  asm volatile("global_load_async_to_lds_b128 %0, %1, off"
               :: "v"(lds_addr), "v"((unsigned long long)(size_t)gaddr)
               : "memory");
}
DEVINL void wait_async0() { asm volatile("s_wait_asynccnt 0x0" ::: "memory"); }
DEVINL unsigned lds_off32(const void* p) { return (unsigned)(size_t)p; }

// ---------------------------------------------------------------------------
// Generic per-wave MLP layer: 16 rows (N = lane&15), MOUT outputs, KIN inputs.
// lds_in:  [16][in_stride] bf16 (in_stride multiple of 16)
// Wt:      global bf16, transposed [MOUT][KIN] (KIN contiguous)
// Writes relu(bf16) activations to lds_out [16][out_stride].
// ---------------------------------------------------------------------------
template<int KIN, int MOUT, bool RELU>
DEVINL void wave_mlp_layer(const unsigned short* lds_in, int in_stride,
                           const unsigned short* __restrict__ Wt,
                           const float* __restrict__ bias,
                           unsigned short* lds_out, int out_stride,
                           int l16, int hp)
{
  v16bf bfr[KIN/32];
#pragma unroll
  for (int kc = 0; kc < KIN/32; ++kc)
    bfr[kc] = ld_v16bf(lds_in + l16*in_stride + kc*32 + hp*16);
#pragma unroll
  for (int mt = 0; mt < MOUT/16; ++mt) {
    v16bf afr[KIN/32];
#pragma unroll
    for (int kc = 0; kc < KIN/32; ++kc)
      afr[kc] = ld_v16bf(Wt + (size_t)(mt*16 + l16)*KIN + kc*32 + hp*16);
    v8f acc;
#pragma unroll
    for (int r = 0; r < 8; ++r) acc[r] = bias[mt*16 + hp*8 + r];
#pragma unroll
    for (int kc = 0; kc < KIN/32; ++kc)
      acc = WMMA_BF16(afr[kc], bfr[kc], acc);
    unsigned pj[4];
#pragma unroll
    for (int j = 0; j < 4; ++j) {
      float v0 = acc[2*j], v1 = acc[2*j+1];
      if (RELU) { v0 = fmaxf(v0, 0.f); v1 = fmaxf(v1, 0.f); }
      pj[j] = (unsigned)f2bfb(v0) | ((unsigned)f2bfb(v1) << 16);
    }
    *(uint4*)(lds_out + l16*out_stride + mt*16 + hp*8) =
        make_uint4(pj[0], pj[1], pj[2], pj[3]);
  }
}

// Same but leaves fp32 accumulators for a custom epilogue.
template<int KIN, int MOUT>
DEVINL void wave_mlp_accum(const unsigned short* lds_in, int in_stride,
                           const unsigned short* __restrict__ Wt,
                           const float* __restrict__ bias,
                           v8f* accs, int l16, int hp)
{
  v16bf bfr[KIN/32];
#pragma unroll
  for (int kc = 0; kc < KIN/32; ++kc)
    bfr[kc] = ld_v16bf(lds_in + l16*in_stride + kc*32 + hp*16);
#pragma unroll
  for (int mt = 0; mt < MOUT/16; ++mt) {
    v16bf afr[KIN/32];
#pragma unroll
    for (int kc = 0; kc < KIN/32; ++kc)
      afr[kc] = ld_v16bf(Wt + (size_t)(mt*16 + l16)*KIN + kc*32 + hp*16);
    v8f acc;
#pragma unroll
    for (int r = 0; r < 8; ++r) acc[r] = bias[mt*16 + hp*8 + r];
#pragma unroll
    for (int kc = 0; kc < KIN/32; ++kc)
      acc = WMMA_BF16(afr[kc], bfr[kc], acc);
    accs[mt] = acc;
  }
}

// ---------------------------------------------------------------------------
// Prep: weight fp32 [in][out] -> bf16 transposed [out][in]
// ---------------------------------------------------------------------------
__global__ __launch_bounds__(256) void k_prep_w(const float* __restrict__ W,
                                                unsigned short* __restrict__ dst,
                                                int in_dim, int out_dim)
{
  int idx = blockIdx.x*256 + threadIdx.x;
  if (idx >= in_dim*out_dim) return;
  int o = idx / in_dim, i = idx - o*in_dim;
  dst[idx] = f2bfb(W[(size_t)i*out_dim + o]);
}

// xs_bf[b][p][0:128] = bf16(cat(x,s))
__global__ __launch_bounds__(256) void k_prep_xs(const float* __restrict__ x,
                                                 const float* __restrict__ s,
                                                 unsigned short* __restrict__ xs_bf)
{
  int i = blockIdx.x*256 + threadIdx.x;
  if (i >= CB*CNX*128) return;
  int c = i & 127, bp = i >> 7;
  float v = (c < 64) ? x[(size_t)bp*64 + c] : s[(size_t)bp*64 + (c - 64)];
  xs_bf[i] = f2bfb(v);
}

__global__ __launch_bounds__(256) void k_zero(float* __restrict__ p, int n)
{
  int i = blockIdx.x*256 + threadIdx.x;
  if (i < n) p[i] = 0.f;
}

// ---------------------------------------------------------------------------
// Encode: latents[b][m][d] = sum_p scores_x[b][p][m] * xs[b][p][d]
// Block = 128 threads (4 waves); block computes 64 M-rows x 128 d.
// ---------------------------------------------------------------------------
__global__ __launch_bounds__(128) void k_encode(const float* __restrict__ scores_x,
                                                const unsigned short* __restrict__ xs_bf,
                                                float* __restrict__ latents,
                                                unsigned short* __restrict__ lat_bf)
{
  __shared__ unsigned short lds_a[64*48];   // [m][k] bf16, stride 48
  __shared__ unsigned short lds_b[128*48];  // [d][k] bf16, stride 48
  const int b = blockIdx.y, mblk = blockIdx.x;
  const int tid = threadIdx.x, lane = tid & 31, wave = tid >> 5;
  const int l16 = lane & 15, hp = lane >> 4;

  v8f acc[8];
#pragma unroll
  for (int n = 0; n < 8; ++n)
#pragma unroll
    for (int r = 0; r < 8; ++r) acc[n][r] = 0.f;

  const float* sx = scores_x + ((size_t)b*CNX)*CM + mblk*64;
  const unsigned short* xb = xs_bf + ((size_t)b*CNX)*128;

  for (int k0 = 0; k0 < CNX; k0 += 32) {
    // stage A slice (transpose fp32 -> bf16): 32 k-rows x 64 m-cols
    for (int t = tid; t < 32*64; t += 128) {
      int kk = t >> 6, mm = t & 63;
      lds_a[mm*48 + kk] = f2bfb(sx[(size_t)(k0 + kk)*CM + mm]);
      if (k0 + 32 < CNX)
        __builtin_prefetch(sx + (size_t)(k0 + 32 + kk)*CM + mm, 0, 0);
    }
    // stage B slice (transpose bf16): 32 k-rows x 128 d (64 dwords/row)
    for (int t = tid; t < 32*64; t += 128) {
      int kk = t >> 6, cu = t & 63;
      unsigned v = ((const unsigned*)(xb + (size_t)(k0 + kk)*128))[cu];
      lds_b[(2*cu    )*48 + kk] = (unsigned short)(v & 0xffffu);
      lds_b[(2*cu + 1)*48 + kk] = (unsigned short)(v >> 16);
    }
    __syncthreads();
    v16bf a = ld_v16bf(lds_a + (wave*16 + l16)*48 + hp*16);
    v16bf bb[8];
#pragma unroll
    for (int nt = 0; nt < 8; ++nt)
      bb[nt] = ld_v16bf(lds_b + (nt*16 + l16)*48 + hp*16);
#pragma unroll
    for (int nt = 0; nt < 8; ++nt)          // 8 independent WMMAs back-to-back
      acc[nt] = WMMA_BF16(a, bb[nt], acc[nt]);
    __syncthreads();
  }
  const size_t base = (size_t)b*CM + mblk*64 + wave*16 + hp*8;
#pragma unroll
  for (int nt = 0; nt < 8; ++nt) {
    int d = nt*16 + l16;
#pragma unroll
    for (int r = 0; r < 8; ++r) {
      float v = acc[nt][r];
      latents[(base + r)*128 + d] = v;
      lat_bf [(base + r)*128 + d] = f2bfb(v);
    }
  }
}

// ---------------------------------------------------------------------------
// Edge messages: async-gather cat(lat[recv], lat[send]) into LDS ->
// 3-layer WMMA MLP -> atomic scatter-add into inbox[recv]. Wave owns 16 edges.
// ---------------------------------------------------------------------------
__global__ __launch_bounds__(128) void k_edge_msg(
    const unsigned short* __restrict__ lat_bf,
    const int* __restrict__ senders, const int* __restrict__ receivers,
    const unsigned short* __restrict__ W0t, const float* __restrict__ b0,
    const unsigned short* __restrict__ W1t, const float* __restrict__ b1,
    const unsigned short* __restrict__ W2t, const float* __restrict__ b2,
    float* __restrict__ inbox)
{
  __shared__ unsigned short s_in[4][16*272];  // stride 272 = 16*17 (544 B/row)
  __shared__ unsigned short s_h [4][16*144];  // stride 144 = 16*9
  const int b = blockIdx.y;
  const int tid = threadIdx.x, lane = tid & 31, wave = tid >> 5;
  const int l16 = lane & 15, hp = lane >> 4;
  const int eBase = blockIdx.x*64 + wave*16;
  unsigned short* lin = s_in[wave];
  unsigned short* lh  = s_h[wave];
  const unsigned short* latb = lat_bf + ((size_t)b*CM)*128;
  const unsigned linA = lds_off32(lin);

  // Async gather: per row, lanes 0-15 copy receiver row (bytes [0,256)),
  // lanes 16-31 copy sender row (bytes [256,512)). 16 B per lane.
#pragma unroll 4
  for (int n = 0; n < 16; ++n) {
    int e = eBase + n;
    int idx = hp ? senders[e] : receivers[e];
    const unsigned short* src = latb + (size_t)idx*128 + l16*8;
    async_ld_b128(linA + n*544 + hp*256 + l16*16, src);
  }
  wait_async0();

  wave_mlp_layer<256,128,true>(lin, 272, W0t, b0, lh, 144, l16, hp);
  wave_mlp_layer<128,128,true>(lh, 144, W1t, b1, lin, 272, l16, hp);
  v8f accs[8];
  wave_mlp_accum<128,128>(lin, 272, W2t, b2, accs, l16, hp);

  const int rc = receivers[eBase + l16];
  float* dst = inbox + ((size_t)b*CM + rc)*128;
#pragma unroll
  for (int mt = 0; mt < 8; ++mt)
#pragma unroll
    for (int r = 0; r < 8; ++r)
      atomicAdd(dst + mt*16 + hp*8 + r, accs[mt][r]);
}

// ---------------------------------------------------------------------------
// Node update: latents += MLP(cat(latents, inbox)); refresh bf16 shadow.
// ---------------------------------------------------------------------------
__global__ __launch_bounds__(128) void k_node_upd(
    float* __restrict__ latents, unsigned short* __restrict__ lat_bf,
    const float* __restrict__ inbox,
    const unsigned short* __restrict__ W0t, const float* __restrict__ b0,
    const unsigned short* __restrict__ W1t, const float* __restrict__ b1,
    const unsigned short* __restrict__ W2t, const float* __restrict__ b2)
{
  __shared__ unsigned short s_in[4][16*272];
  __shared__ unsigned short s_h [4][16*144];
  const int b = blockIdx.y;
  const int tid = threadIdx.x, lane = tid & 31, wave = tid >> 5;
  const int l16 = lane & 15, hp = lane >> 4;
  const int nodeBase = blockIdx.x*64 + wave*16;
  unsigned short* lin = s_in[wave];
  unsigned short* lh  = s_h[wave];
  const unsigned short* latb = lat_bf + ((size_t)b*CM)*128;
  const float* inb = inbox + ((size_t)b*CM)*128;
  const unsigned linA = lds_off32(lin);

  // latents half (bytes [0,256) of each row): async copy, 2 rows per op
#pragma unroll
  for (int n = 0; n < 16; n += 2) {
    int row = nodeBase + n + hp;     // lanes 0-15 -> row n, 16-31 -> row n+1
    const unsigned short* src = latb + (size_t)row*128 + l16*8;
    async_ld_b128(linA + (n + hp)*544 + l16*16, src);
  }
  // inbox half (fp32 -> bf16 convert): manual
  for (int n = 0; n < 16; ++n) {
    const float* ir = inb + (size_t)(nodeBase + n)*128;
    for (int c = lane; c < 128; c += 32) lin[n*272 + 128 + c] = f2bfb(ir[c]);
  }
  wait_async0();

  wave_mlp_layer<256,128,true>(lin, 272, W0t, b0, lh, 144, l16, hp);
  wave_mlp_layer<128,128,true>(lh, 144, W1t, b1, lin, 272, l16, hp);
  v8f accs[8];
  wave_mlp_accum<128,128>(lin, 272, W2t, b2, accs, l16, hp);

  const int row = nodeBase + l16;
  float* lrow = latents + ((size_t)b*CM + row)*128;
  unsigned short* brow = lat_bf + ((size_t)b*CM + row)*128;
#pragma unroll
  for (int mt = 0; mt < 8; ++mt)
#pragma unroll
    for (int r = 0; r < 8; ++r) {
      int d = mt*16 + hp*8 + r;
      float v = lrow[d] + accs[mt][r];
      lrow[d] = v;
      brow[d] = f2bfb(v);
    }
}

// ---------------------------------------------------------------------------
// Decode GEMM: z[b][q][d] = sum_m scores_q[b][q][m] * latents[b][m][d]
// A (scores_q) rows are K-contiguous -> direct global loads + cvt.
// ---------------------------------------------------------------------------
__global__ __launch_bounds__(128) void k_decode_gemm(
    const float* __restrict__ scores_q,
    const unsigned short* __restrict__ lat_bf,
    unsigned short* __restrict__ z_bf)
{
  __shared__ unsigned short lds_b[128*48];  // [d][k] bf16
  const int b = blockIdx.y, qblk = blockIdx.x;
  const int tid = threadIdx.x, lane = tid & 31, wave = tid >> 5;
  const int l16 = lane & 15, hp = lane >> 4;

  v8f acc[8];
#pragma unroll
  for (int n = 0; n < 8; ++n)
#pragma unroll
    for (int r = 0; r < 8; ++r) acc[n][r] = 0.f;

  const float* sq = scores_q + ((size_t)b*CNQ + qblk*64 + wave*16 + l16)*CM;
  const unsigned short* lb = lat_bf + ((size_t)b*CM)*128;

  union { v16bf v; unsigned short u[16]; } A;

  for (int k0 = 0; k0 < CM; k0 += 32) {
    for (int t = tid; t < 32*64; t += 128) {
      int kk = t >> 6, cu = t & 63;
      unsigned v = ((const unsigned*)(lb + (size_t)(k0 + kk)*128))[cu];
      lds_b[(2*cu    )*48 + kk] = (unsigned short)(v & 0xffffu);
      lds_b[(2*cu + 1)*48 + kk] = (unsigned short)(v >> 16);
    }
    __syncthreads();
    const float4* ap = (const float4*)(sq + k0 + hp*16);
    float fa[16];
    *(float4*)&fa[0]  = ap[0];
    *(float4*)&fa[4]  = ap[1];
    *(float4*)&fa[8]  = ap[2];
    *(float4*)&fa[12] = ap[3];
    if (k0 + 32 < CM) __builtin_prefetch(sq + k0 + 32 + hp*16, 0, 0);
#pragma unroll
    for (int j = 0; j < 16; ++j) A.u[j] = f2bfb(fa[j]);
    v16bf bb[8];
#pragma unroll
    for (int nt = 0; nt < 8; ++nt)
      bb[nt] = ld_v16bf(lds_b + (nt*16 + l16)*48 + hp*16);
#pragma unroll
    for (int nt = 0; nt < 8; ++nt)
      acc[nt] = WMMA_BF16(A.v, bb[nt], acc[nt]);
    __syncthreads();
  }
  const size_t qbase = (size_t)b*CNQ + qblk*64 + wave*16 + hp*8;
#pragma unroll
  for (int nt = 0; nt < 8; ++nt)
#pragma unroll
    for (int r = 0; r < 8; ++r)
      z_bf[(qbase + r)*128 + nt*16 + l16] = f2bfb(acc[nt][r]);
}

// ---------------------------------------------------------------------------
// Decoder MLP: out = MLP(cat(z, q)), dims 192 -> 128 -> 128 -> 64 (fp32 out)
// ---------------------------------------------------------------------------
__global__ __launch_bounds__(128) void k_dec_mlp(
    const unsigned short* __restrict__ z_bf, const float* __restrict__ q,
    const unsigned short* __restrict__ W0t, const float* __restrict__ b0,
    const unsigned short* __restrict__ W1t, const float* __restrict__ b1,
    const unsigned short* __restrict__ W2t, const float* __restrict__ b2,
    float* __restrict__ out)
{
  __shared__ unsigned short s_in[4][16*208];  // 208 = 16*13 (416 B/row)
  __shared__ unsigned short s_h [4][16*144];
  const int b = blockIdx.y;
  const int tid = threadIdx.x, lane = tid & 31, wave = tid >> 5;
  const int l16 = lane & 15, hp = lane >> 4;
  const int rowBase = blockIdx.x*64 + wave*16;
  unsigned short* lin = s_in[wave];
  unsigned short* lh  = s_h[wave];
  const unsigned linA = lds_off32(lin);

  // z half (bytes [0,256)): async copy, 2 rows per op
#pragma unroll
  for (int n = 0; n < 16; n += 2) {
    int row = rowBase + n + hp;
    const unsigned short* src = z_bf + ((size_t)b*CNQ + row)*128 + l16*8;
    async_ld_b128(linA + (n + hp)*416 + l16*16, src);
  }
  // q half (fp32 -> bf16)
  for (int n = 0; n < 16; ++n) {
    const float* qr = q + ((size_t)b*CNQ + rowBase + n)*64;
    for (int c = lane; c < 64; c += 32) lin[n*208 + 128 + c] = f2bfb(qr[c]);
  }
  wait_async0();

  wave_mlp_layer<192,128,true>(lin, 208, W0t, b0, lh, 144, l16, hp);
  wave_mlp_layer<128,128,true>(lh, 144, W1t, b1, lin, 208, l16, hp);
  v8f accs[4];
  wave_mlp_accum<128,64>(lin, 208, W2t, b2, accs, l16, hp);

  const int row = rowBase + l16;
  float* orow = out + ((size_t)b*CNQ + row)*64;
#pragma unroll
  for (int mt = 0; mt < 4; ++mt)
#pragma unroll
    for (int r = 0; r < 8; ++r)
      orow[mt*16 + hp*8 + r] = accs[mt][r];
}

// ---------------------------------------------------------------------------
// Host orchestration
// ---------------------------------------------------------------------------
extern "C" void kernel_launch(void* const* d_in, const int* in_sizes, int n_in,
                              void* d_out, int out_size, void* d_ws, size_t ws_size,
                              hipStream_t stream)
{
  const float* x        = (const float*)d_in[0];
  const float* s        = (const float*)d_in[1];
  const float* q        = (const float*)d_in[2];
  const float* scores_x = (const float*)d_in[3];
  const float* scores_q = (const float*)d_in[4];
  const int*   senders   = (const int*)d_in[5];
  const int*   receivers = (const int*)d_in[6];

  // workspace layout (bytes)
  char* w = (char*)d_ws;
  float* latents = (float*)(w + 0);                         //  8 MB
  float* inbox   = (float*)(w + 8388608);                   //  8 MB
  unsigned short* lat_bf = (unsigned short*)(w + 16777216); //  4 MB
  unsigned short* xs_bf  = (unsigned short*)(w + 20971520); //  1 MB
  unsigned short* z_bf   = (unsigned short*)(w + 22020096); //  1 MB
  unsigned short* wb     = (unsigned short*)(w + 23068672); // ~0.9 MB bf16 weights

  auto prepW = [&](int idx, unsigned short* dst, int in_dim, int out_dim) {
    int n = in_dim*out_dim;
    k_prep_w<<<dim3((n + 255)/256), dim3(256), 0, stream>>>(
        (const float*)d_in[idx], dst, in_dim, out_dim);
  };

  // blocks[step] = (msg MLP, node MLP); each MLP = 3x(W,b); params start at 7
  for (int st = 0; st < 3; ++st) {
    int base = 7 + st*12;
    unsigned short* so = wb + st*131072;
    prepW(base + 0,  so + 0,      256, 128);  // msg W0
    prepW(base + 2,  so + 32768,  128, 128);  // msg W1
    prepW(base + 4,  so + 49152,  128, 128);  // msg W2
    prepW(base + 6,  so + 65536,  256, 128);  // node W0
    prepW(base + 8,  so + 98304,  128, 128);  // node W1
    prepW(base + 10, so + 114688, 128, 128);  // node W2
  }
  prepW(43, wb + 393216, 192, 128);  // dec W0
  prepW(45, wb + 417792, 128, 128);  // dec W1
  prepW(47, wb + 434176, 128, 64);   // dec W2

  k_prep_xs<<<dim3((CB*CNX*128 + 255)/256), dim3(256), 0, stream>>>(x, s, xs_bf);

  k_encode<<<dim3(CM/64, CB), dim3(128), 0, stream>>>(scores_x, xs_bf, latents, lat_bf);

  const int nInbox = CB*CM*128;
  for (int st = 0; st < 3; ++st) {
    int base = 7 + st*12;
    unsigned short* so = wb + st*131072;
    k_zero<<<dim3((nInbox + 255)/256), dim3(256), 0, stream>>>(inbox, nInbox);
    k_edge_msg<<<dim3(CE/64, CB), dim3(128), 0, stream>>>(
        lat_bf, senders, receivers,
        so + 0,      (const float*)d_in[base + 1],
        so + 32768,  (const float*)d_in[base + 3],
        so + 49152,  (const float*)d_in[base + 5],
        inbox);
    k_node_upd<<<dim3(CM/64, CB), dim3(128), 0, stream>>>(
        latents, lat_bf, inbox,
        so + 65536,  (const float*)d_in[base + 7],
        so + 98304,  (const float*)d_in[base + 9],
        so + 114688, (const float*)d_in[base + 11]);
  }

  k_decode_gemm<<<dim3(CNQ/64, CB), dim3(128), 0, stream>>>(scores_q, lat_bf, z_bf);

  k_dec_mlp<<<dim3(CNQ/64, CB), dim3(128), 0, stream>>>(
      z_bf, q,
      wb + 393216, (const float*)d_in[44],
      wb + 417792, (const float*)d_in[46],
      wb + 434176, (const float*)d_in[48],
      (float*)d_out);
}
